// JoraLayer_88313117540371
// MI455X (gfx1250) — compile-verified, matrix operand order checked by reading, not run on hardware
//
#include <hip/hip_runtime.h>
#include <hip/hip_bf16.h>
#include <stddef.h>

// ---------------------------------------------------------------------------
// JoRA layer for MI455X (gfx1250, wave32, WMMA).
//   out = x @ W_eff^T + bias,  W_eff = rowscale(rot_rows(rot_cols(W)))
// Strategy: tiny f32 preprocessing (memory-bound), then bf16x3 split-precision
// GEMM using v_wmma_f32_16x16x32_bf16 (hh + hl + lh terms ~ fp32 accuracy).
// ---------------------------------------------------------------------------

typedef __attribute__((ext_vector_type(16))) __bf16        v16bf;
typedef __attribute__((ext_vector_type(8)))  float         v8f;
typedef __attribute__((ext_vector_type(4)))  unsigned int  u32x4;

union FragAB {
    u32x4 u[2];   // two 16-byte chunks (8 bf16 each)
    v16bf v;      // 16 bf16 = 8 VGPRs, WMMA A/B fragment
};

__device__ __forceinline__ unsigned short f2bf(float f) {
    unsigned int u = __float_as_uint(f);
    u += 0x7FFFu + ((u >> 16) & 1u);        // round-to-nearest-even
    return (unsigned short)(u >> 16);
}
__device__ __forceinline__ float bf2f(unsigned short h) {
    return __uint_as_float(((unsigned int)h) << 16);
}

// ---------------- preprocessing kernels (all memory-bound, tiny) -----------

__global__ void k_sincos(const float* __restrict__ th, float* __restrict__ cs, int S) {
    int i = blockIdx.x * blockDim.x + threadIdx.x;
    if (i < S) {
        cs[2 * i + 0] = cosf(th[i]);
        cs[2 * i + 1] = sinf(th[i]);
    }
}

// Column-pair rotation. One block per row; row staged in LDS so the
// scattered pair gathers are LDS hits instead of strided HBM reads.
__global__ void k_colrot(const float* __restrict__ W, float* __restrict__ W1,
                         const int* __restrict__ pairs, const float* __restrict__ cs,
                         int S, int M) {
    extern __shared__ float srow[];
    int r = blockIdx.x;
    const float* wr = W + (size_t)r * M;
    for (int c = threadIdx.x; c < M; c += blockDim.x) srow[c] = wr[c];
    __syncthreads();
    for (int p = threadIdx.x; p < S; p += blockDim.x) {
        int i = pairs[2 * p], j = pairs[2 * p + 1];
        float c = cs[2 * p], s = cs[2 * p + 1];
        float ci = srow[i], cj = srow[j];          // disjoint pairs -> in-place safe
        srow[i] = c * ci - s * cj;
        srow[j] = s * ci + c * cj;
    }
    __syncthreads();
    float* orow = W1 + (size_t)r * M;
    for (int c = threadIdx.x; c < M; c += blockDim.x) orow[c] = srow[c];
}

// Row-pair rotation (in place, disjoint pairs -> each block owns its 2 rows)
// fused with row sum-of-squares reduction.
__global__ void k_rowrot(float* __restrict__ W1, const int* __restrict__ pairs,
                         const float* __restrict__ cs, float* __restrict__ row_ssq,
                         int M) {
    __shared__ float redI[256], redJ[256];
    int p = blockIdx.x;
    int i = pairs[2 * p], j = pairs[2 * p + 1];
    float c = cs[2 * p], s = cs[2 * p + 1];
    float* ri = W1 + (size_t)i * M;
    float* rj = W1 + (size_t)j * M;
    float si = 0.f, sj = 0.f;
    for (int col = threadIdx.x; col < M; col += blockDim.x) {
        float a = ri[col], b = rj[col];
        float wi = c * a - s * b;
        float wj = s * a + c * b;
        ri[col] = wi; rj[col] = wj;
        si += wi * wi; sj += wj * wj;
    }
    redI[threadIdx.x] = si; redJ[threadIdx.x] = sj;
    __syncthreads();
    for (int st = blockDim.x >> 1; st > 0; st >>= 1) {
        if ((int)threadIdx.x < st) {
            redI[threadIdx.x] += redI[threadIdx.x + st];
            redJ[threadIdx.x] += redJ[threadIdx.x + st];
        }
        __syncthreads();
    }
    if (threadIdx.x == 0) { row_ssq[i] = redI[0]; row_ssq[j] = redJ[0]; }
}

__global__ void k_scale(const float* __restrict__ brn, const float* __restrict__ ecd,
                        const float* __restrict__ ssq, float* __restrict__ scale, int N) {
    int r = blockIdx.x * blockDim.x + threadIdx.x;
    if (r < N) scale[r] = brn[r] * expf(ecd[r]) * rsqrtf(ssq[r] + 1e-6f);
}

// W_eff = W1 * scale[row], split into bf16 hi/lo planes (row-major N x M).
__global__ void k_splitW(const float* __restrict__ W1, const float* __restrict__ scale,
                         unsigned short* __restrict__ wh, unsigned short* __restrict__ wl,
                         int M) {
    int r = blockIdx.x;
    float sc = scale[r];
    const float* src = W1 + (size_t)r * M;
    unsigned short* ph = wh + (size_t)r * M;
    unsigned short* pl = wl + (size_t)r * M;
    for (int c = threadIdx.x; c < M; c += blockDim.x) {
        float w = src[c] * sc;
        unsigned short h = f2bf(w);
        ph[c] = h;
        pl[c] = f2bf(w - bf2f(h));
    }
}

__global__ void k_splitX(const float* __restrict__ x, unsigned short* __restrict__ xh,
                         unsigned short* __restrict__ xl, size_t n) {
    size_t i = (size_t)blockIdx.x * blockDim.x + threadIdx.x;
    if (i < n) {
        float f = x[i];
        unsigned short h = f2bf(f);
        xh[i] = h;
        xl[i] = f2bf(f - bf2f(h));
    }
}

// ---------------- main GEMM: out[T,N] = x @ W_eff^T + bias -----------------
// Block = 256 threads = 8 waves; block tile 128(M) x 128(N).
// Wave tile 64(M) x 32(N) = 4x2 WMMA 16x16 tiles; per K=32 step: 24 b128
// loads + 24 v_wmma_f32_16x16x32_bf16 (terms hh, hl, lh).
//
// Fragment load pattern follows the ISA bf16 A/B layout:
//   lane&15 -> matrix row (A: M, B: N), lane>>4 selects K half (+8),
//   two 16B chunks per fragment at K and K+16.
__global__ __launch_bounds__(256)
void k_gemm_bf16x3(const unsigned short* __restrict__ xh,
                   const unsigned short* __restrict__ xl,
                   const unsigned short* __restrict__ wh,
                   const unsigned short* __restrict__ wl,
                   const float* __restrict__ bias,
                   float* __restrict__ out,
                   int T, int N, int M) {
    const int lane   = threadIdx.x & 31;
    const int wave   = threadIdx.x >> 5;
    const int waveM  = wave >> 2;                 // 0..1
    const int waveN  = wave & 3;                  // 0..3
    const int mBase  = blockIdx.y * 128 + waveM * 64;
    const int nBase  = blockIdx.x * 128 + waveN * 32;
    const int lane16 = lane & 15;
    const int halfK  = (lane >> 4) * 8;           // 0 or 8 (bf16 elements)

    v8f acc[4][2];
    #pragma unroll
    for (int mi = 0; mi < 4; ++mi)
        #pragma unroll
        for (int ni = 0; ni < 2; ++ni)
            #pragma unroll
            for (int e = 0; e < 8; ++e) acc[mi][ni][e] = 0.0f;

    size_t aoff[4], boff[2];
    #pragma unroll
    for (int mi = 0; mi < 4; ++mi)
        aoff[mi] = (size_t)(mBase + mi * 16 + lane16) * (size_t)M + halfK;
    #pragma unroll
    for (int ni = 0; ni < 2; ++ni)
        boff[ni] = (size_t)(nBase + ni * 16 + lane16) * (size_t)M + halfK;

    for (int k = 0; k < M; k += 32) {
        FragAB ah[4], al[4], bh[2], bl[2];
        #pragma unroll
        for (int mi = 0; mi < 4; ++mi) {
            const unsigned short* p = xh + aoff[mi] + k;
            ah[mi].u[0] = *(const u32x4*)(p);
            ah[mi].u[1] = *(const u32x4*)(p + 16);
            const unsigned short* q = xl + aoff[mi] + k;
            al[mi].u[0] = *(const u32x4*)(q);
            al[mi].u[1] = *(const u32x4*)(q + 16);
        }
        #pragma unroll
        for (int ni = 0; ni < 2; ++ni) {
            const unsigned short* p = wh + boff[ni] + k;
            bh[ni].u[0] = *(const u32x4*)(p);
            bh[ni].u[1] = *(const u32x4*)(p + 16);
            const unsigned short* q = wl + boff[ni] + k;
            bl[ni].u[0] = *(const u32x4*)(q);
            bl[ni].u[1] = *(const u32x4*)(q + 16);
        }
        #pragma unroll
        for (int mi = 0; mi < 4; ++mi) {
            #pragma unroll
            for (int ni = 0; ni < 2; ++ni) {
                acc[mi][ni] = __builtin_amdgcn_wmma_f32_16x16x32_bf16(
                    false, ah[mi].v, false, bh[ni].v, (short)0, acc[mi][ni], false, false);
                acc[mi][ni] = __builtin_amdgcn_wmma_f32_16x16x32_bf16(
                    false, ah[mi].v, false, bl[ni].v, (short)0, acc[mi][ni], false, false);
                acc[mi][ni] = __builtin_amdgcn_wmma_f32_16x16x32_bf16(
                    false, al[mi].v, false, bh[ni].v, (short)0, acc[mi][ni], false, false);
            }
        }
    }

    // Epilogue: C/D layout -> lane = N col, VGPR e = M row (lanes>=16: +8).
    #pragma unroll
    for (int ni = 0; ni < 2; ++ni) {
        int n = nBase + ni * 16 + lane16;
        float bv = bias[n];
        #pragma unroll
        for (int mi = 0; mi < 4; ++mi) {
            int mrow = mBase + mi * 16 + halfK;    // halfK doubles as M offset
            #pragma unroll
            for (int e = 0; e < 8; ++e) {
                out[(size_t)(mrow + e) * (size_t)N + n] = acc[mi][ni][e] + bv;
            }
        }
    }
}

// ---------------------------------------------------------------------------

extern "C" void kernel_launch(void* const* d_in, const int* in_sizes, int n_in,
                              void* d_out, int out_size, void* d_ws, size_t ws_size,
                              hipStream_t stream) {
    const float* x    = (const float*)d_in[0];
    const float* W    = (const float*)d_in[1];
    const float* bias = (const float*)d_in[2];
    const float* thL  = (const float*)d_in[3];
    const float* thR  = (const float*)d_in[4];
    const float* ecd  = (const float*)d_in[5];
    const float* brn  = (const float*)d_in[6];
    const int*   pL   = (const int*)d_in[7];
    const int*   pR   = (const int*)d_in[8];

    const int N  = in_sizes[2];
    const int SL = in_sizes[3];
    const int SR = in_sizes[4];
    const int M  = in_sizes[1] / N;
    const int T  = in_sizes[0] / M;

    float* outp = (float*)d_out;

    // Workspace layout (256B-aligned slices).
    char* ws = (char*)d_ws;
    size_t off = 0;
    auto take = [&](size_t bytes) {
        void* p = ws + off;
        off += (bytes + 255) & ~(size_t)255;
        return p;
    };
    float*          W1      = (float*)take((size_t)N * M * sizeof(float));
    float*          csR     = (float*)take((size_t)SR * 2 * sizeof(float));
    float*          csL     = (float*)take((size_t)SL * 2 * sizeof(float));
    float*          row_ssq = (float*)take((size_t)N * sizeof(float));
    float*          scale   = (float*)take((size_t)N * sizeof(float));
    unsigned short* whp     = (unsigned short*)take((size_t)N * M * sizeof(unsigned short));
    unsigned short* wlp     = (unsigned short*)take((size_t)N * M * sizeof(unsigned short));
    unsigned short* xhp     = (unsigned short*)take((size_t)T * M * sizeof(unsigned short));
    unsigned short* xlp     = (unsigned short*)take((size_t)T * M * sizeof(unsigned short));
    (void)ws_size; (void)n_in; (void)out_size;

    k_sincos<<<(SR + 255) / 256, 256, 0, stream>>>(thR, csR, SR);
    k_sincos<<<(SL + 255) / 256, 256, 0, stream>>>(thL, csL, SL);
    k_colrot<<<N, 256, (size_t)M * sizeof(float), stream>>>(W, W1, pR, csR, SR, M);
    k_rowrot<<<SL, 256, 0, stream>>>(W1, pL, csL, row_ssq, M);
    k_scale<<<(N + 255) / 256, 256, 0, stream>>>(brn, ecd, row_ssq, scale, N);
    k_splitW<<<N, 256, 0, stream>>>(W1, scale, whp, wlp, M);
    size_t nx = (size_t)T * M;
    k_splitX<<<(unsigned)((nx + 255) / 256), 256, 0, stream>>>(x, xhp, xlp, nx);

    dim3 grid(N / 128, T / 128);
    k_gemm_bf16x3<<<grid, 256, 0, stream>>>(xhp, xlp, whp, wlp, bias, outp, T, N, M);
}
